// MeshUnpool_7464653160711
// MI455X (gfx1250) — matrix operand order repaired and checked by reading, not compiled, not run on hardware
//
#include <hip/hip_runtime.h>

typedef __attribute__((ext_vector_type(16))) _Float16 v16h;
typedef __attribute__((ext_vector_type(8)))  _Float16 v8h;
typedef __attribute__((ext_vector_type(8)))  float    v8f;
typedef __attribute__((ext_vector_type(4)))  float    v4f;

// Problem dims (fixed by the reference)
constexpr int NB = 16;    // batch
constexpr int NF = 256;   // rows of A (features)
constexpr int KE = 2048;  // reduction dim (edges)
constexpr int NT = 4096;  // cols of B (unroll targets)

// Blocking
constexpr int BM = 128;
constexpr int BN = 128;
constexpr int BK = 64;
constexpr int KIT = KE / BK;   // 32 K-iterations
constexpr int LDR = 72;        // LDS row pitch in halves (64 payload + 8 pad -> 144B, 16B aligned)

__global__ __launch_bounds__(256) void
meshunpool_wmma_f16(const float* __restrict__ features,
                    const float* __restrict__ groups,
                    const float* __restrict__ occ,
                    float* __restrict__ out)
{
    __shared__ _Float16 As[2][BM * LDR];   // [buf][m][k]  row-major
    __shared__ _Float16 Bs[2][BN * LDR];   // [buf][n][k]  transposed (column of B contiguous)

    const int tid = threadIdx.x;
    const int w   = tid >> 5;        // wave 0..7
    const int l   = tid & 31;        // lane
    const int lo  = l & 15;
    const int hi  = l >> 4;
    const int wm  = w >> 2;          // 0..1 : wave M position
    const int wn  = w & 3;           // 0..3 : wave N position

    const int bid = blockIdx.x;      // 1024 blocks: nb fastest (A reuse in L2)
    const int nb  = bid & 31;
    const int mb  = (bid >> 5) & 1;
    const int b   = bid >> 6;
    const int m0  = mb * BM;
    const int n0  = nb * BN;

    const float* Fb = features + (size_t)b * NF * KE;
    const float* Gb = groups   + (size_t)b * KE * NT;
    const float* Ob = occ      + (size_t)b * NT;

    // Per-lane reciprocal occurrences for the 4 B columns this lane stages (n = l + 32*i)
    float ro[4];
#pragma unroll
    for (int i = 0; i < 4; ++i)
        ro[i] = 1.0f / Ob[n0 + l + 32 * i];

    // --- staging assignments ---
    // A: thread stages row am, k-half ak (32 floats) -> 8x v4f coalesced-ish loads, 4x ds_store_b128
    const int am = tid >> 1;
    const int ak = (tid & 1) * 32;
    const float* gA = Fb + (size_t)(m0 + am) * KE + ak;
    // B: wave w stages rows k = 8w..8w+7; lane covers columns l, l+32, l+64, l+96 (coalesced b32)
    const int bk = w * 8;
    const float* gB = Gb + (size_t)bk * NT + n0 + l;

    float a_st[32];
    float b_st[32];

    auto gload = [&](int it) {
        const int k0 = it * BK;
#pragma unroll
        for (int i = 0; i < 8; ++i) {
            v4f v = *(const v4f*)(gA + k0 + i * 4);
#pragma unroll
            for (int e = 0; e < 4; ++e) a_st[i * 4 + e] = v[e];
        }
        const float* gb = gB + (size_t)k0 * NT;
#pragma unroll
        for (int r = 0; r < 8; ++r)
#pragma unroll
            for (int i = 0; i < 4; ++i)
                b_st[r * 4 + i] = __builtin_nontemporal_load(gb + (size_t)r * NT + 32 * i);
    };

    auto sstore = [&](int buf) {
        _Float16* Ar = &As[buf][am * LDR + ak];
#pragma unroll
        for (int j = 0; j < 4; ++j) {
            v8h h;
#pragma unroll
            for (int e = 0; e < 8; ++e) h[e] = (_Float16)a_st[j * 8 + e];
            *(v8h*)(Ar + j * 8) = h;            // ds_store_b128
        }
#pragma unroll
        for (int i = 0; i < 4; ++i) {
            _Float16* Br = &Bs[buf][(l + 32 * i) * LDR + bk];
#pragma unroll
            for (int r = 0; r < 8; ++r)         // fused 1/occ + f16 cvt, transposed scatter
                Br[r] = (_Float16)(b_st[r * 4 + i] * ro[i]);
        }
    };

    v8f acc[4][2] = {};

    auto compute = [&](int buf) {
#pragma unroll
        for (int f = 0; f < 2; ++f) {           // two 16x16x32 K-fragments per BK=64
            v16h afr[4];
            v16h bfr[2];
#pragma unroll
            for (int mt = 0; mt < 4; ++mt) {
                // A layout: lane(hi) holds K = hi*16+0..15 (contiguous, matches B)
                const _Float16* ap = &As[buf][(wm * 64 + mt * 16 + lo) * LDR + f * 32];
                v8h x = *(const v8h*)(ap + hi * 16);
                v8h y = *(const v8h*)(ap + hi * 16 + 8);
                afr[mt] = __builtin_shufflevector(x, y, 0, 1, 2, 3, 4, 5, 6, 7,
                                                         8, 9, 10, 11, 12, 13, 14, 15);
            }
#pragma unroll
            for (int nt = 0; nt < 2; ++nt) {
                // B layout: lanes 0-15 hold K=0..15, lanes 16-31 hold K=16..31 (contiguous in Bt)
                const _Float16* bp = &Bs[buf][(wn * 32 + nt * 16 + lo) * LDR + f * 32 + hi * 16];
                v8h x = *(const v8h*)(bp);
                v8h y = *(const v8h*)(bp + 8);
                bfr[nt] = __builtin_shufflevector(x, y, 0, 1, 2, 3, 4, 5, 6, 7,
                                                        8, 9, 10, 11, 12, 13, 14, 15);
            }
#pragma unroll
            for (int mt = 0; mt < 4; ++mt)
#pragma unroll
                for (int nt = 0; nt < 2; ++nt)
                    acc[mt][nt] = __builtin_amdgcn_wmma_f32_16x16x32_f16(
                        false, afr[mt], false, bfr[nt],
                        (short)0, acc[mt][nt], false, false);
        }
    };

    // --- pipelined main loop: prefetch globals for it+1 while computing it ---
    gload(0);
    sstore(0);
    __syncthreads();
    for (int it = 0; it < KIT; ++it) {
        if (it + 1 < KIT) gload(it + 1);
        compute(it & 1);
        if (it + 1 < KIT) sstore((it + 1) & 1);
        __syncthreads();
    }

    // --- epilogue: C/D layout -> M = v + hi*8, N = lo ---
    float* Cp = out + (size_t)b * NF * NT;
#pragma unroll
    for (int mt = 0; mt < 4; ++mt) {
        const int mrow = m0 + wm * 64 + mt * 16 + hi * 8;
#pragma unroll
        for (int nt = 0; nt < 2; ++nt) {
            const int ncol = n0 + wn * 32 + nt * 16 + lo;
#pragma unroll
            for (int v = 0; v < 8; ++v)
                __builtin_nontemporal_store(acc[mt][nt][v],
                    Cp + (size_t)(mrow + v) * NT + ncol);
        }
    }
}

extern "C" void kernel_launch(void* const* d_in, const int* in_sizes, int n_in,
                              void* d_out, int out_size, void* d_ws, size_t ws_size,
                              hipStream_t stream)
{
    const float* features = (const float*)d_in[0];   // [16, 256, 2048] f32
    const float* groups   = (const float*)d_in[1];   // [16, 2048, 4096] f32
    const float* occ      = (const float*)d_in[2];   // [16, 4096] f32
    float*       out      = (float*)d_out;           // [16, 256, 4096] f32

    dim3 grid(NB * (NF / BM) * (NT / BN));           // 16 * 2 * 32 = 1024
    dim3 block(256);
    meshunpool_wmma_f16<<<grid, block, 0, stream>>>(features, groups, occ, out);
}
